// MoE_84361747628174
// MI455X (gfx1250) — matrix operand order repaired and checked by reading, not compiled, not run on hardware
//
#include <hip/hip_runtime.h>
#include <hip/hip_bf16.h>
#include <math.h>

#define T_TOK 8192
#define DIM   1024
#define NEXP  16
#define ESZ   256
#define TOPK  4

typedef __attribute__((ext_vector_type(16))) __bf16 v16bf;
typedef __attribute__((ext_vector_type(8)))  __bf16 v8bf;
typedef __attribute__((ext_vector_type(8)))  float  v8f;

__device__ __forceinline__ unsigned short f2bf(float f){
    unsigned u = __float_as_uint(f);
    u += 0x7FFFu + ((u >> 16) & 1u);           // round-to-nearest-even
    return (unsigned short)(u >> 16);
}

// ---------------- zero out + counters ----------------
__global__ void zero_kernel(float* __restrict__ out, int n, int* __restrict__ counts){
    int i = blockIdx.x * blockDim.x + threadIdx.x;
    int stride = gridDim.x * blockDim.x;
    for (; i < n; i += stride) out[i] = 0.0f;
    if (blockIdx.x == 0 && threadIdx.x < NEXP) counts[threadIdx.x] = 0;
}

// ---------------- x -> bf16 (packed pairs) ----------------
__global__ void cvt_x_kernel(const float* __restrict__ x, unsigned* __restrict__ xb, int n2){
    int i = blockIdx.x * blockDim.x + threadIdx.x;
    int stride = gridDim.x * blockDim.x;
    for (; i < n2; i += stride){
        float a = x[2*i+0], b = x[2*i+1];
        xb[i] = (unsigned)f2bf(a) | ((unsigned)f2bf(b) << 16);
    }
}

// ---------------- weights: [E][R][C] f32 -> [E][C][R] bf16 (N-major for WMMA B) ----
__global__ void tr_cvt_kernel(const float* __restrict__ in, unsigned short* __restrict__ outp,
                              int R, int C){
    __shared__ unsigned short tile[32][33];
    int e = blockIdx.x, r0 = blockIdx.y * 32, c0 = blockIdx.z * 32;
    const float* ine = in + (size_t)e * R * C;
    unsigned short* oute = outp + (size_t)e * R * C;
    int tr = threadIdx.x >> 5, tc = threadIdx.x & 31;
    #pragma unroll
    for (int i = 0; i < 4; ++i){
        int r = tr + i * 8;
        tile[r][tc] = f2bf(ine[(size_t)(r0 + r) * C + (c0 + tc)]);
    }
    __syncthreads();
    #pragma unroll
    for (int i = 0; i < 4; ++i){
        int c = tr + i * 8;
        oute[(size_t)(c0 + c) * R + (r0 + tc)] = tile[tc][c];
    }
}

// ---------------- fp32 gating + top-4 + per-expert token lists ----------------
__global__ void gate_kernel(const float* __restrict__ x, const float* __restrict__ wg,
                            int* __restrict__ counts, int* __restrict__ tok_list,
                            float* __restrict__ gate_list){
    extern __shared__ float wgs[];   // NEXP*DIM floats = 64KB
    for (int i = threadIdx.x; i < NEXP*DIM; i += blockDim.x) wgs[i] = wg[i];
    __syncthreads();
    int t = blockIdx.x * blockDim.x + threadIdx.x;
    const float* xr = x + (size_t)t * DIM;
    float acc[NEXP];
    #pragma unroll
    for (int e = 0; e < NEXP; ++e) acc[e] = 0.0f;
    for (int d = 0; d < DIM; ++d){
        float xv = xr[d];
        #pragma unroll
        for (int e = 0; e < NEXP; ++e) acc[e] = fmaf(xv, wgs[e*DIM + d], acc[e]);
    }
    #pragma unroll
    for (int e = 0; e < NEXP; ++e) acc[e] = 1.0f / (1.0f + __expf(-acc[e]));
    #pragma unroll
    for (int k = 0; k < TOPK; ++k){
        float best = -1.0f; int bi = 0;
        #pragma unroll
        for (int e = 0; e < NEXP; ++e){           // strict > keeps lowest index on ties (matches top_k)
            if (acc[e] > best){ best = acc[e]; bi = e; }
        }
        acc[bi] = -2.0f;
        int pos = atomicAdd(&counts[bi], 1);
        tok_list[bi*T_TOK + pos]  = t;
        gate_list[bi*T_TOK + pos] = best;
    }
}

// ---------------- expert FFN: gather -> WMMA -> scatter-add ----------------
#define MT 64     // tokens per block
#define KC 64     // K-chunk for keys pass
#define NC 64     // N-chunk for values pass

// LDS map (bytes), single dynamic allocation (base 0):
//   HS        [64][256] bf16               @ 0      .. 32768
//   stage buf0: XT[64][64] + KT[256][64]   @ 32768  .. 73728
//   stage buf1: XT[64][64] + KT[256][64]   @ 73728  .. 114688
//   (values phase VT[64][256] x2 alias the stage region @32768/@65536)
//   tok[64], gate[64]                      @ 114688 .. 115200
#define STG0_OFF   32768
#define STG_STRIDE 40960
#define VT_STRIDE  32768
#define TOK_OFF    114688
#define GATE_OFF   114944
#define SMEM_BYTES 115200

__device__ __forceinline__ void async_b128(unsigned lds_off, unsigned goff, const void* base){
    // GVS mode: per-lane 32-bit byte offset + uniform 64-bit SGPR base; tracked by ASYNCcnt
    asm volatile("global_load_async_to_lds_b128 %0, %1, %2"
                 :: "v"(lds_off), "v"(goff), "s"(base) : "memory");
}
__device__ __forceinline__ void wait_async(){
    asm volatile("s_wait_asynccnt 0x0" ::: "memory");
}

__global__ __launch_bounds__(256) void expert_kernel(
    const unsigned short* __restrict__ xbf,   // [T][D] bf16
    const unsigned short* __restrict__ kT,    // [E][ES][D] bf16 (N-major keys)
    const unsigned short* __restrict__ vT,    // [E][D][ES] bf16 (N-major values)
    const int* __restrict__ counts,
    const int* __restrict__ tok_list,
    const float* __restrict__ gate_list,
    float* __restrict__ out)
{
    extern __shared__ char smem[];
    const int e  = blockIdx.x;
    const int tb = blockIdx.y;
    const int cnt = counts[e];
    if (tb * MT >= cnt) return;

    const int tid  = threadIdx.x;
    const int wave = tid >> 5;
    const int lane = tid & 31;
    const int hi   = lane >> 4;   // lane half (ISA fragment layouts)
    const int l4   = lane & 15;

    int*   tok_s  = (int*)(smem + TOK_OFF);
    float* gate_s = (float*)(smem + GATE_OFF);
    if (tid < MT){
        int g = tb * MT + tid;
        if (g < cnt){
            tok_s[tid]  = tok_list[e * T_TOK + g];
            gate_s[tid] = gate_list[e * T_TOK + g];
        } else { tok_s[tid] = 0; gate_s[tid] = 0.0f; }
    }
    __syncthreads();

    const unsigned short* kTe = kT + (size_t)e * ESZ * DIM;
    const unsigned short* vTe = vT + (size_t)e * DIM * ESZ;

    const int mt    = wave >> 1;     // 0..3 : 16-row block
    const int nhalf = wave & 1;      // 0..1 : N half

    const v8f vzero = {0.f,0.f,0.f,0.f,0.f,0.f,0.f,0.f};

    // ================= keys pass: H[64,256] = relu(Xg[64,1024] @ Ke) =================
    const int NBASE = nhalf * 128;
    v8f c[8];
    #pragma unroll
    for (int i = 0; i < 8; ++i) c[i] = vzero;

    // staging helper (inlined twice): X tile [64][KC] + Ke^T tile [256][KC]
    auto stage_keys = [&](int kc, int buf){
        unsigned sbase = (unsigned)(STG0_OFF + buf * STG_STRIDE);
        #pragma unroll
        for (int it = 0; it < 2; ++it){
            int idx = tid + it * 256;
            int row = idx >> 3, seg = idx & 7;
            unsigned goff = (unsigned)((tok_s[row] * DIM + kc + seg * 8) * 2);
            async_b128(sbase + (unsigned)((row * KC + seg * 8) * 2), goff, xbf);
        }
        #pragma unroll
        for (int it = 0; it < 8; ++it){
            int idx = tid + it * 256;
            int row = idx >> 3, seg = idx & 7;
            unsigned goff = (unsigned)((row * DIM + kc + seg * 8) * 2);
            async_b128(sbase + 8192u + (unsigned)((row * KC + seg * 8) * 2), goff, kTe);
        }
    };

    stage_keys(0, 0);
    wait_async();
    __syncthreads();

    const int NCH_K = DIM / KC;   // 16
    for (int ch = 0; ch < NCH_K; ++ch){
        int buf = ch & 1;
        if (ch + 1 < NCH_K) stage_keys((ch + 1) * KC, buf ^ 1);   // prefetch next chunk (async)

        unsigned xb = (unsigned)(STG0_OFF + buf * STG_STRIDE);
        unsigned kb = xb + 8192u;
        #pragma unroll
        for (int ks = 0; ks < KC; ks += 32){
            int arow = mt * 16 + l4;
            int r0 = ks + hi * 8;                               // A 16x32 bf16 lane layout
            v8bf alo = *(const v8bf*)(smem + xb + (arow * KC + r0) * 2);
            v8bf ahi = *(const v8bf*)(smem + xb + (arow * KC + r0 + 16) * 2);
            v16bf a = __builtin_shufflevector(alo, ahi, 0,1,2,3,4,5,6,7,8,9,10,11,12,13,14,15);
            v16bf bfr[8];
            #pragma unroll
            for (int nt = 0; nt < 8; ++nt){                     // batch all B loads first
                int n   = NBASE + nt * 16 + l4;                 // B 32x16: lane holds column n
                int kst = ks + hi * 8;
                v8bf blo = *(const v8bf*)(smem + kb + (n * KC + kst) * 2);
                v8bf bhi = *(const v8bf*)(smem + kb + (n * KC + kst + 16) * 2);
                bfr[nt] = __builtin_shufflevector(blo, bhi, 0,1,2,3,4,5,6,7,8,9,10,11,12,13,14,15);
            }
            #pragma unroll
            for (int nt = 0; nt < 8; ++nt)                      // back-to-back WMMAs
                c[nt] = __builtin_amdgcn_wmma_f32_16x16x32_bf16(false, a, false, bfr[nt],
                                                                (short)0, c[nt], false, false);
        }
        wait_async();
        __syncthreads();
    }

    // ---- relu * gate -> HS bf16 (A-operand layout for values pass) ----
    #pragma unroll
    for (int nt = 0; nt < 8; ++nt){
        #pragma unroll
        for (int j = 0; j < 8; ++j){
            int m = mt * 16 + hi * 8 + j;                       // C/D layout: M = j + 8*hi
            int n = NBASE + nt * 16 + l4;
            float v = fmaxf(c[nt][j], 0.0f) * gate_s[m];
            *(unsigned short*)(smem + (m * ESZ + n) * 2) = f2bf(v);
        }
    }
    __syncthreads();

    // ================= values pass: O[64,1024] = (g.H)[64,256] @ Ve =================
    auto stage_vals = [&](int nb, int buf){
        unsigned vbase = (unsigned)(STG0_OFF + buf * VT_STRIDE);
        #pragma unroll
        for (int it = 0; it < 8; ++it){                         // Ve^T tile [NC=64][256]
            int idx = tid + it * 256;
            int row = idx >> 5, seg = idx & 31;
            unsigned goff = (unsigned)(((nb + row) * ESZ + seg * 8) * 2);
            async_b128(vbase + (unsigned)((row * ESZ + seg * 8) * 2), goff, vTe);
        }
    };

    stage_vals(0, 0);

    // Hoist the 8 A-fragments (HS rows for this wave) into registers for the whole pass.
    v16bf a_arr[8];
    {
        int arow = mt * 16 + l4;
        #pragma unroll
        for (int k8 = 0; k8 < 8; ++k8){
            int r0 = k8 * 32 + hi * 8;
            v8bf alo = *(const v8bf*)(smem + (arow * ESZ + r0) * 2);
            v8bf ahi = *(const v8bf*)(smem + (arow * ESZ + r0 + 16) * 2);
            a_arr[k8] = __builtin_shufflevector(alo, ahi, 0,1,2,3,4,5,6,7,8,9,10,11,12,13,14,15);
        }
    }
    wait_async();
    __syncthreads();

    const int NCH_V = DIM / NC;   // 16
    for (int ch = 0; ch < NCH_V; ++ch){
        int buf = ch & 1;
        if (ch + 1 < NCH_V) stage_vals((ch + 1) * NC, buf ^ 1); // prefetch next chunk (async)

        unsigned vb = (unsigned)(STG0_OFF + buf * VT_STRIDE);
        v8f o[2];
        o[0] = vzero; o[1] = vzero;

        #pragma unroll
        for (int k8 = 0; k8 < 8; ++k8){
            int kst = k8 * 32 + hi * 8;
            v16bf bfr[2];
            #pragma unroll
            for (int nt = 0; nt < 2; ++nt){
                int vr = (nhalf * 2 + nt) * 16 + l4;            // local column in this 64-wide chunk
                v8bf blo = *(const v8bf*)(smem + vb + (vr * ESZ + kst) * 2);
                v8bf bhi = *(const v8bf*)(smem + vb + (vr * ESZ + kst + 16) * 2);
                bfr[nt] = __builtin_shufflevector(blo, bhi, 0,1,2,3,4,5,6,7,8,9,10,11,12,13,14,15);
            }
            #pragma unroll
            for (int nt = 0; nt < 2; ++nt)
                o[nt] = __builtin_amdgcn_wmma_f32_16x16x32_bf16(false, a_arr[k8], false, bfr[nt],
                                                                (short)0, o[nt], false, false);
        }

        int nb = ch * NC;
        #pragma unroll
        for (int nt = 0; nt < 2; ++nt){
            #pragma unroll
            for (int j = 0; j < 8; ++j){
                int m = mt * 16 + hi * 8 + j;
                if (tb * MT + m < cnt){
                    int n = nb + (nhalf * 2 + nt) * 16 + l4;
                    atomicAdd(&out[(size_t)tok_s[m] * DIM + n], o[nt][j]);
                }
            }
        }
        wait_async();
        __syncthreads();
    }
}

extern "C" void kernel_launch(void* const* d_in, const int* in_sizes, int n_in,
                              void* d_out, int out_size, void* d_ws, size_t ws_size,
                              hipStream_t stream){
    const float* x      = (const float*)d_in[0];
    const float* w_gate = (const float*)d_in[1];
    const float* keys   = (const float*)d_in[2];
    const float* values = (const float*)d_in[3];
    float* out = (float*)d_out;

    char* ws = (char*)d_ws;
    // layout: counts(256B) | tok_list 512KB | gate_list 512KB | xbf 16MB | kT 8MB | vT 8MB  (~33MB)
    int*            counts    = (int*)(ws + 0);
    int*            tok_list  = (int*)(ws + 256);
    float*          gate_list = (float*)(ws + 256 + 524288);
    unsigned short* xbf       = (unsigned short*)(ws + 1048832);
    unsigned short* kT        = (unsigned short*)(ws + 17826048);
    unsigned short* vT        = (unsigned short*)(ws + 26214656);

    zero_kernel <<<2048, 256, 0, stream>>>(out, T_TOK*DIM, counts);
    cvt_x_kernel<<<2048, 256, 0, stream>>>(x, (unsigned*)xbf, T_TOK*DIM/2);
    tr_cvt_kernel<<<dim3(NEXP, DIM/32, ESZ/32), 256, 0, stream>>>(keys,   kT, DIM, ESZ);
    tr_cvt_kernel<<<dim3(NEXP, ESZ/32, DIM/32), 256, 0, stream>>>(values, vT, ESZ, DIM);
    gate_kernel <<<T_TOK/256, 256, NEXP*DIM*4, stream>>>(x, w_gate, counts, tok_list, gate_list);
    expert_kernel<<<dim3(NEXP, T_TOK/MT), 256, SMEM_BYTES, stream>>>(
        xbf, kT, vT, counts, tok_list, gate_list, out);
}
